// Model_22170621182330
// MI455X (gfx1250) — compile-verified, matrix operand order checked by reading, not run on hardware
//
#include <hip/hip_runtime.h>

typedef __attribute__((ext_vector_type(16))) _Float16 v16h;
typedef __attribute__((ext_vector_type(8)))  _Float16 v8h;
typedef __attribute__((ext_vector_type(8)))  float    v8f;

#define DEVFN static __device__ __forceinline__

// ---- WMMA fragment helpers (16x16x32 f16 -> f32) -------------------------
// ISA 16-bit A layout (16x32 MxK): lane<16 holds row lane, lane>=16 row-16.
// Element e of the v16h maps to K = (e>=8?16:0) + (lane>>4)*8 + (e&7), i.e.
// per lane TWO contiguous 8-half runs -> two 16B vector loads.
DEVFN v16h frag_a(const _Float16* p, int row0, int k0, int ld) {
  int lane = threadIdx.x & 31;
  const _Float16* base = p + (row0 + (lane & 15)) * ld + k0 + ((lane >> 4) << 3);
  v8h lo = *(const v8h*)base;         // K = hb .. hb+7
  v8h hi = *(const v8h*)(base + 16);  // K = 16+hb .. 16+hb+7
  return __builtin_shufflevector(lo, hi, 0, 1, 2, 3, 4, 5, 6, 7,
                                 8, 9, 10, 11, 12, 13, 14, 15);
}

// B fragments come from weights pre-packed as [kt][nt][lane][16]: one
// contiguous 32B load per lane.
DEVFN v16h frag_b_pk(const _Float16* p, int kt, int nt, int ntN) {
  int lane = threadIdx.x & 31;
  return *(const v16h*)(p + (((kt * ntN + nt) << 9) | (lane << 4)));
}

DEVFN v8f wmma16(v16h a, v16h b, v8f c) {
  return __builtin_amdgcn_wmma_f32_16x16x32_f16(false, a, false, b,
                                                (short)0, c, false, false);
}
// C/D layout: lane holds column (lane&15); VGPR i holds row i + (lane>>4)*8.

// ---- weight prep: f32 [K,N] row-major -> f16 fragment-packed tiles -------
// dst element i: b = i/(K*N); tile = (i%(K*N))>>9; lane=(i>>4)&31; e=i&15.
// K may exceed Ksrc (zero-padded K rows).
__global__ void pack_b(const float* __restrict__ src, _Float16* __restrict__ dst,
                       int K, int N, int Ksrc, int nbatch, int srcStride) {
  int i = blockIdx.x * blockDim.x + threadIdx.x;
  int perB = K * N;
  if (i >= nbatch * perB) return;
  int b = i / perB;
  int r = i % perB;
  int tile = r >> 9;
  int lane = (r >> 4) & 31;
  int e = r & 15;
  int ntN = N >> 4;
  int kt = tile / ntN, nt = tile % ntN;
  int n = nt * 16 + (lane & 15);
  int k = kt * 32 + ((e & 8) ? 16 : 0) + ((lane >> 4) << 3) + (e & 7);
  float v = (k < Ksrc) ? src[b * srcStride + k * N + n] : 0.f;
  dst[i] = (_Float16)v;
}

// ---- K1: move encoder ----------------------------------------------------
// 16 move-tokens (= 4 pokes) per workgroup; WMMA 16x32 @ 32x64.
__global__ void move_enc(const int* __restrict__ moveInts, const float* __restrict__ moveFeats,
                         const float* __restrict__ moveEmb, const float* __restrict__ typeEmb,
                         const float* __restrict__ U, const float* __restrict__ Wf,
                         const float* __restrict__ ln_g, const float* __restrict__ ln_b,
                         const _Float16* __restrict__ w1, const float* __restrict__ b1,
                         _Float16* __restrict__ movesOut) {
  __shared__ float sm[16][32];
  __shared__ float mu_s[16], rs_s[16];
  __shared__ alignas(16) _Float16 aF[16][32];
  int lane = threadIdx.x;
  int tile = blockIdx.x;

  for (int t = 0; t < 16; ++t) {           // lane owns feature column `lane`
    int mt = tile * 16 + t;
    int i0 = moveInts[mt * 2 + 0];
    int i1 = moveInts[mt * 2 + 1];
    float v = moveEmb[i0 * 32 + lane];
#pragma unroll
    for (int j = 0; j < 10; ++j) v += typeEmb[i1 * 10 + j] * U[j * 32 + lane];
#pragma unroll
    for (int j = 0; j < 6; ++j)  v += moveFeats[mt * 6 + j] * Wf[j * 32 + lane];
    sm[t][lane] = v;
  }
  __syncthreads();
  if (lane < 16) {
    float m = 0.f;
    for (int f = 0; f < 32; ++f) m += sm[lane][f];
    m *= (1.f / 32.f);
    float var = 0.f;
    for (int f = 0; f < 32; ++f) { float d = sm[lane][f] - m; var += d * d; }
    mu_s[lane] = m;
    rs_s[lane] = rsqrtf(var * (1.f / 32.f) + 1e-5f);
  }
  __syncthreads();
  {
    float g = ln_g[lane], bb = ln_b[lane];
    for (int t = 0; t < 16; ++t)
      aF[t][lane] = (_Float16)((sm[t][lane] - mu_s[t]) * rs_s[t] * g + bb);
  }
  __syncthreads();

  v16h a = frag_a(&aF[0][0], 0, 0, 32);
  int hi = lane >> 4, col16 = lane & 15;
  for (int nt = 0; nt < 4; ++nt) {
    v8f c = {};
    c = wmma16(a, frag_b_pk(w1, 0, nt, 4), c);
    int col = nt * 16 + col16;
    float bias = b1[col];
    float s0 = 0.f, s1 = 0.f;
#pragma unroll
    for (int i = 0; i < 4; ++i) s0 += fmaxf(c[i] + bias, 0.f);
#pragma unroll
    for (int i = 4; i < 8; ++i) s1 += fmaxf(c[i] + bias, 0.f);
    int pokeBase = tile * 4 + hi * 2;      // rows 0-3,4-7,8-11,12-15 = pokes
    movesOut[pokeBase * 64 + col]       = (_Float16)s0;
    movesOut[(pokeBase + 1) * 64 + col] = (_Float16)s1;
  }
}

// ---- K2: poke encoder ----------------------------------------------------
// 16 poke-tokens per workgroup; WMMA 16x224 @ 224x128 (K padded 215->224).
__global__ void poke_enc(const int* __restrict__ pokeInts, const float* __restrict__ pokeFeats,
                         const float* __restrict__ pokeEmb, const float* __restrict__ itemEmb,
                         const float* __restrict__ abEmb,  const float* __restrict__ typeEmb,
                         const _Float16* __restrict__ moves,
                         const float* __restrict__ ln_g, const float* __restrict__ ln_b,
                         const _Float16* __restrict__ wlin, const float* __restrict__ blin,
                         float* __restrict__ xOut) {
  __shared__ float sf[16 * 224];
  __shared__ alignas(16) _Float16 aF[16 * 224];
  __shared__ float mu_s[16], rs_s[16];
  int lane = threadIdx.x;
  int tile = blockIdx.x;

  for (int idx = lane; idx < 16 * 224; idx += 32) {
    int t = idx / 224, c = idx % 224;
    int tok = tile * 16 + t;
    const int* pi = pokeInts + tok * 6;
    float v = 0.f;
    if (c < 32)        v = pokeEmb[pi[0] * 32 + c];
    else if (c < 64)   v = itemEmb[pi[1] * 32 + (c - 32)];
    else if (c < 96)   v = abEmb[pi[2] * 32 + (c - 64)];
    else if (c < 126) { int cc = c - 96; v = typeEmb[pi[3 + cc / 10] * 10 + cc % 10]; }
    else if (c < 151)  v = pokeFeats[tok * 25 + (c - 126)];
    else if (c < 215)  v = (float)moves[tok * 64 + (c - 151)];
    sf[idx] = v;
  }
  __syncthreads();
  if (lane < 16) {
    const float* row = sf + lane * 224;
    float m = 0.f;
    for (int j = 0; j < 215; ++j) m += row[j];
    m *= (1.f / 215.f);
    float var = 0.f;
    for (int j = 0; j < 215; ++j) { float d = row[j] - m; var += d * d; }
    mu_s[lane] = m;
    rs_s[lane] = rsqrtf(var * (1.f / 215.f) + 1e-5f);
  }
  __syncthreads();
  for (int idx = lane; idx < 16 * 224; idx += 32) {
    int t = idx / 224, c = idx % 224;
    float v = 0.f;
    if (c < 215) v = (sf[idx] - mu_s[t]) * rs_s[t] * ln_g[c] + ln_b[c];
    aF[idx] = (_Float16)v;
  }
  __syncthreads();

  int hi = lane >> 4, col16 = lane & 15;
  for (int nt = 0; nt < 8; ++nt) {
    v8f c = {};
#pragma unroll
    for (int kt = 0; kt < 7; ++kt)
      c = wmma16(frag_a(aF, 0, kt * 32, 224), frag_b_pk(wlin, kt, nt, 8), c);
    int col = nt * 16 + col16;
    float bias = blin[col];
#pragma unroll
    for (int i = 0; i < 8; ++i) {
      int row = i + hi * 8;
      xOut[(size_t)(tile * 16 + row) * 128 + col] = fmaxf(c[i] + bias, 0.f);
    }
  }
}

// ---- K3: pre-LN MHA block (one batch item per workgroup) -----------------
__global__ void mha_block(float* __restrict__ x,
                          const float* __restrict__ ln_g, const float* __restrict__ ln_b,
                          const _Float16* __restrict__ wk, const _Float16* __restrict__ wq,
                          const _Float16* __restrict__ wv,
                          const _Float16* __restrict__ wproj, const float* __restrict__ bproj) {
  __shared__ float xf[16 * 128];
  __shared__ alignas(16) _Float16 xn[16 * 128];
  __shared__ float qs[16 * 16], ks[16 * 16], vs[16 * 16];
  __shared__ float sc[16 * 16];
  __shared__ alignas(16) _Float16 hd[16 * 128];
  __shared__ float mu_s[16], rs_s[16];
  int lane = threadIdx.x;
  float* xg = x + (size_t)blockIdx.x * 12 * 128;
  __builtin_prefetch(xg + lane * 32, 0, 1);   // global_prefetch_b8

  for (int idx = lane; idx < 16 * 128; idx += 32) {
    xf[idx] = (idx < 12 * 128) ? xg[idx] : 0.f;
    hd[idx] = (_Float16)0.f;
  }
  __syncthreads();
  if (lane < 16) {
    const float* row = xf + lane * 128;
    float m = 0.f;
    for (int j = 0; j < 128; ++j) m += row[j];
    m *= (1.f / 128.f);
    float var = 0.f;
    for (int j = 0; j < 128; ++j) { float d = row[j] - m; var += d * d; }
    mu_s[lane] = m;
    rs_s[lane] = rsqrtf(var * (1.f / 128.f) + 1e-5f);
  }
  __syncthreads();
  for (int idx = lane; idx < 16 * 128; idx += 32) {
    int t = idx >> 7, cc = idx & 127;
    float v = (t < 12) ? (xf[idx] - mu_s[t]) * rs_s[t] * ln_g[cc] + ln_b[cc] : 0.f;
    xn[idx] = (_Float16)v;
  }
  __syncthreads();

  int hi = lane >> 4, col16 = lane & 15;
  for (int h = 0; h < 8; ++h) {
    const _Float16* wqh = wq + h * 2048;   // packed [4 kt][1 nt][32][16]
    const _Float16* wkh = wk + h * 2048;
    const _Float16* wvh = wv + h * 2048;
    v8f cq = {}, ck = {}, cv = {};
#pragma unroll
    for (int kt = 0; kt < 4; ++kt) {
      v16h a = frag_a(xn, 0, kt * 32, 128);
      cq = wmma16(a, frag_b_pk(wqh, kt, 0, 1), cq);
      ck = wmma16(a, frag_b_pk(wkh, kt, 0, 1), ck);
      cv = wmma16(a, frag_b_pk(wvh, kt, 0, 1), cv);
    }
#pragma unroll
    for (int i = 0; i < 8; ++i) {
      int row = i + hi * 8;
      qs[row * 16 + col16] = cq[i];
      ks[row * 16 + col16] = ck[i];
      vs[row * 16 + col16] = cv[i];
    }
    __syncthreads();
    for (int idx = lane; idx < 144; idx += 32) {
      int s = idx / 12, t = idx % 12;
      float d = 0.f;
#pragma unroll
      for (int e = 0; e < 16; ++e) d += qs[s * 16 + e] * ks[t * 16 + e];
      sc[s * 16 + t] = d * 0.25f;          // 1/sqrt(16)
    }
    __syncthreads();
    if (lane < 12) {
      float mx = -1e30f;
      for (int t = 0; t < 12; ++t) mx = fmaxf(mx, sc[lane * 16 + t]);
      float sum = 0.f;
      for (int t = 0; t < 12; ++t) { float e = __expf(sc[lane * 16 + t] - mx); sc[lane * 16 + t] = e; sum += e; }
      float inv = 1.f / sum;
      for (int t = 0; t < 12; ++t) sc[lane * 16 + t] *= inv;
    }
    __syncthreads();
    for (int idx = lane; idx < 192; idx += 32) {
      int s = idx / 16, e = idx % 16;
      float v = 0.f;
      for (int t = 0; t < 12; ++t) v += sc[s * 16 + t] * vs[t * 16 + e];
      hd[s * 128 + h * 16 + e] = (_Float16)v;
    }
    __syncthreads();
  }
  // proj + residual (in-place)
  for (int nt = 0; nt < 8; ++nt) {
    v8f c = {};
#pragma unroll
    for (int kt = 0; kt < 4; ++kt)
      c = wmma16(frag_a(hd, 0, kt * 32, 128), frag_b_pk(wproj, kt, nt, 8), c);
    int col = nt * 16 + col16;
    float bias = bproj[col];
#pragma unroll
    for (int i = 0; i < 8; ++i) {
      int row = i + hi * 8;
      if (row < 12) xg[row * 128 + col] = xf[row * 128 + col] + c[i] + bias;
    }
  }
}

// ---- K4: attention pooling + board encoder + bn1 -> h[B,288] f16 ---------
__global__ void pool_head(const float* __restrict__ x,
                          const int* __restrict__ boardInts, const float* __restrict__ boardFeats,
                          const float* __restrict__ twEmb, const float* __restrict__ trEmb,
                          const float* __restrict__ weEmb, const float* __restrict__ teEmb,
                          const float* __restrict__ bg, const float* __restrict__ bb,
                          const float* __restrict__ bW, const float* __restrict__ bbias,
                          const float* __restrict__ poolW, const float* __restrict__ poolB,
                          const float* __restrict__ bn1g, const float* __restrict__ bn1b,
                          _Float16* __restrict__ hOut) {
  __shared__ float xl[12 * 128];
  __shared__ float wts[12];
  __shared__ float comb[35], nrm[35];
  __shared__ float board[32];
  __shared__ float pooled[256];
  int lane = threadIdx.x;
  int b = blockIdx.x;
  const float* xg = x + (size_t)b * 1536;
  __builtin_prefetch(xg + lane * 32, 0, 1);
  for (int idx = lane; idx < 1536; idx += 32) xl[idx] = xg[idx];

  const int* bi = boardInts + b * 5;
  for (int idx = lane; idx < 35; idx += 32) {
    float v;
    if (idx < 4)       v = twEmb[bi[0] * 4 + idx];
    else if (idx < 8)  v = twEmb[bi[1] * 4 + idx - 4];
    else if (idx < 12) v = trEmb[bi[2] * 4 + idx - 8];
    else if (idx < 16) v = weEmb[bi[3] * 4 + idx - 12];
    else if (idx < 20) v = teEmb[bi[4] * 4 + idx - 16];
    else               v = boardFeats[b * 15 + idx - 20];
    comb[idx] = v;
  }
  __syncthreads();
  if (lane < 12) {                       // pool logits
    float d = poolB[0];
    for (int j = 0; j < 128; ++j) d += xl[lane * 128 + j] * poolW[j];
    wts[lane] = d;
  }
  if (lane == 0) {                       // board LN
    float m = 0.f;
    for (int j = 0; j < 35; ++j) m += comb[j];
    m *= (1.f / 35.f);
    float var = 0.f;
    for (int j = 0; j < 35; ++j) { float d = comb[j] - m; var += d * d; }
    float rs = rsqrtf(var * (1.f / 35.f) + 1e-5f);
    for (int j = 0; j < 35; ++j) nrm[j] = (comb[j] - m) * rs * bg[j] + bb[j];
  }
  __syncthreads();
  if (lane < 2) {                        // softmax over each team of 6
    float mx = -1e30f;
    for (int p = 0; p < 6; ++p) mx = fmaxf(mx, wts[lane * 6 + p]);
    float s = 0.f, e[6];
    for (int p = 0; p < 6; ++p) { e[p] = __expf(wts[lane * 6 + p] - mx); s += e[p]; }
    for (int p = 0; p < 6; ++p) wts[lane * 6 + p] = e[p] / s;
  }
  {                                      // board linear + relu (lane = out col)
    float d = bbias[lane];
    for (int j = 0; j < 35; ++j) d += nrm[j] * bW[j * 32 + lane];
    board[lane] = fmaxf(d, 0.f);
  }
  __syncthreads();
  for (int idx = lane; idx < 256; idx += 32) {
    int team = idx >> 7, cc = idx & 127;
    float v = 0.f;
    for (int p = 0; p < 6; ++p) v += xl[(team * 6 + p) * 128 + cc] * wts[team * 6 + p];
    pooled[idx] = v;
  }
  __syncthreads();
  float inv = rsqrtf(1.f + 1e-5f);       // BatchNorm eval, running stats 0/1
  for (int j = lane; j < 288; j += 32) {
    float v = (j < 256) ? pooled[j] : board[j - 256];
    hOut[(size_t)b * 288 + j] = (_Float16)(v * bn1g[j] * inv + bn1b[j]);
  }
}

// ---- K5: head MLP, 16 items per workgroup --------------------------------
__global__ void head_mlp(const _Float16* __restrict__ h,
                         const _Float16* __restrict__ w1, const float* __restrict__ b1,
                         const float* __restrict__ bn2g, const float* __restrict__ bn2b,
                         const _Float16* __restrict__ w2, const float* __restrict__ b2,
                         const float* __restrict__ wf, const float* __restrict__ bf,
                         float* __restrict__ out) {
  __shared__ alignas(16) _Float16 a2[16 * 64];
  __shared__ float o2[16 * 16];
  int lane = threadIdx.x;
  int tile = blockIdx.x;
  const _Float16* hg = h + (size_t)tile * 16 * 288;
  __builtin_prefetch(hg + lane * 144, 0, 1);
  int hi = lane >> 4, col16 = lane & 15;
  float inv = rsqrtf(1.f + 1e-5f);

  for (int nt = 0; nt < 4; ++nt) {       // fc1: 16x288 @ 288x64
    v8f c1 = {};
#pragma unroll
    for (int kt = 0; kt < 9; ++kt)
      c1 = wmma16(frag_a(hg, 0, kt * 32, 288), frag_b_pk(w1, kt, nt, 4), c1);
    int col = nt * 16 + col16;
    float bias = b1[col], g2 = bn2g[col] * inv, bb2 = bn2b[col];
#pragma unroll
    for (int i = 0; i < 8; ++i) {
      int row = i + hi * 8;
      float v = fmaxf(c1[i] + bias, 0.f);
      a2[row * 64 + col] = (_Float16)(v * g2 + bb2);   // bn2 fused here
    }
  }
  __syncthreads();
  v8f c2 = {};                            // fc2: 16x64 @ 64x16
#pragma unroll
  for (int kt = 0; kt < 2; ++kt)
    c2 = wmma16(frag_a(a2, 0, kt * 32, 64), frag_b_pk(w2, kt, 0, 1), c2);
#pragma unroll
  for (int i = 0; i < 8; ++i) {
    int row = i + hi * 8;
    o2[row * 16 + col16] = fmaxf(c2[i] + b2[col16], 0.f);
  }
  __syncthreads();
  if (lane < 16) {
    float d = bf[0];
    for (int j = 0; j < 16; ++j) d += o2[lane * 16 + j] * wf[j];
    out[tile * 16 + lane] = d;
  }
}

// ---- host launcher -------------------------------------------------------
extern "C" void kernel_launch(void* const* d_in, const int* in_sizes, int n_in,
                              void* d_out, int out_size, void* d_ws, size_t ws_size,
                              hipStream_t stream) {
  (void)n_in; (void)out_size; (void)ws_size;
  const int B = in_sizes[0] / 5;

  const int*   boardInts  = (const int*)d_in[0];
  const float* boardFeats = (const float*)d_in[1];
  const int*   pokeInts   = (const int*)d_in[2];
  const float* pokeFeats  = (const float*)d_in[3];
  const int*   moveInts   = (const int*)d_in[4];
  const float* moveFeats  = (const float*)d_in[5];

  enum { P_twEmb = 6, P_trEmb, P_weatherEmb, P_terrainEmb, P_b_ln_g, P_b_ln_b,
         P_b_lin_W, P_b_lin_b, P_typeEmb, P_moveEmb, P_U, P_W, P_m_ln1_g,
         P_m_ln1_b, P_m_lin1_W, P_m_lin1_b, P_pokeEmb, P_itemEmb, P_abEmb,
         P_p_ln_g, P_p_ln_b, P_p_lin_W, P_p_lin_b, P_blk_ln_g, P_blk_ln_b,
         P_Wk, P_Wq, P_Wv, P_mha_proj_W, P_mha_proj_b, P_pool_W, P_pool_b,
         P_bn1_g, P_bn1_b, P_fc1_W, P_fc1_b, P_bn2_g, P_bn2_b, P_fc2_W,
         P_fc2_b, P_final_W, P_final_b };
  #define PF(i) ((const float*)d_in[(i)])

  char* ws = (char*)d_ws;
  size_t off = 0;
  auto alloc = [&](size_t bytes) -> void* {
    off = (off + 255) & ~(size_t)255;
    void* p = ws + off;
    off += bytes;
    return p;
  };
  _Float16* movesBuf = (_Float16*)alloc((size_t)B * 12 * 64 * 2);
  float*    xBuf     = (float*)   alloc((size_t)B * 12 * 128 * 4);
  _Float16* hBuf     = (_Float16*)alloc((size_t)B * 288 * 2);
  _Float16* w_m1   = (_Float16*)alloc(32 * 64 * 2);
  _Float16* w_plin = (_Float16*)alloc(224 * 128 * 2);
  _Float16* w_k    = (_Float16*)alloc(8 * 128 * 16 * 2);
  _Float16* w_q    = (_Float16*)alloc(8 * 128 * 16 * 2);
  _Float16* w_v    = (_Float16*)alloc(8 * 128 * 16 * 2);
  _Float16* w_proj = (_Float16*)alloc(128 * 128 * 2);
  _Float16* w_fc1  = (_Float16*)alloc(288 * 64 * 2);
  _Float16* w_fc2  = (_Float16*)alloc(64 * 16 * 2);

  auto pack = [&](int pidx, _Float16* dst, int K, int N, int Ksrc,
                  int nbatch, int srcStride) {
    int n = nbatch * K * N;
    pack_b<<<(n + 255) / 256, 256, 0, stream>>>(PF(pidx), dst, K, N, Ksrc,
                                                nbatch, srcStride);
  };
  pack(P_m_lin1_W,   w_m1,   32, 64, 32, 1, 0);
  pack(P_p_lin_W,    w_plin, 224, 128, 215, 1, 0);
  pack(P_Wk,         w_k,    128, 16, 128, 8, 2048);
  pack(P_Wq,         w_q,    128, 16, 128, 8, 2048);
  pack(P_Wv,         w_v,    128, 16, 128, 8, 2048);
  pack(P_mha_proj_W, w_proj, 128, 128, 128, 1, 0);
  pack(P_fc1_W,      w_fc1,  288, 64, 288, 1, 0);
  pack(P_fc2_W,      w_fc2,  64, 16, 64, 1, 0);

  move_enc<<<B * 3, 32, 0, stream>>>(moveInts, moveFeats, PF(P_moveEmb), PF(P_typeEmb),
                                     PF(P_U), PF(P_W), PF(P_m_ln1_g), PF(P_m_ln1_b),
                                     w_m1, PF(P_m_lin1_b), movesBuf);

  poke_enc<<<(B * 12) / 16, 32, 0, stream>>>(pokeInts, pokeFeats, PF(P_pokeEmb),
                                             PF(P_itemEmb), PF(P_abEmb), PF(P_typeEmb),
                                             movesBuf, PF(P_p_ln_g), PF(P_p_ln_b),
                                             w_plin, PF(P_p_lin_b), xBuf);

  mha_block<<<B, 32, 0, stream>>>(xBuf, PF(P_blk_ln_g), PF(P_blk_ln_b),
                                  w_k, w_q, w_v, w_proj, PF(P_mha_proj_b));

  pool_head<<<B, 32, 0, stream>>>(xBuf, boardInts, boardFeats,
                                  PF(P_twEmb), PF(P_trEmb), PF(P_weatherEmb), PF(P_terrainEmb),
                                  PF(P_b_ln_g), PF(P_b_ln_b), PF(P_b_lin_W), PF(P_b_lin_b),
                                  PF(P_pool_W), PF(P_pool_b), PF(P_bn1_g), PF(P_bn1_b), hBuf);

  head_mlp<<<B / 16, 32, 0, stream>>>(hBuf, w_fc1, PF(P_fc1_b), PF(P_bn2_g), PF(P_bn2_b),
                                      w_fc2, PF(P_fc2_b), PF(P_final_W), PF(P_final_b),
                                      (float*)d_out);
  #undef PF
}
